// SubGraph_28346784153784
// MI455X (gfx1250) — matrix operand order repaired and checked by reading, not compiled, not run on hardware
//
#include <hip/hip_runtime.h>

#define NSEG 100000

typedef __attribute__((ext_vector_type(16))) _Float16 v16h;
typedef __attribute__((ext_vector_type(8)))  _Float16 v8h;
typedef __attribute__((ext_vector_type(8)))  float    v8f;

union BFrag { unsigned u[8]; v16h h; };
union AFrag { v8h h8[2]; v16h h16; };
union HPack { _Float16 h[2]; unsigned u; };

// ---- order-preserving float <-> uint key (for atomicMax-based segment max) ----
__device__ __forceinline__ unsigned enc_f32(float f) {
  unsigned b = __float_as_uint(f);
  return (b & 0x80000000u) ? ~b : (b | 0x80000000u);
}
// key(-inf) == 0x007FFFFF is the init value; decode it (empty segment) to 0
__device__ __forceinline__ float dec_key(unsigned k) {
  if (k == 0x007FFFFFu) return 0.0f;
  return (k & 0x80000000u) ? __uint_as_float(k ^ 0x80000000u)
                           : __uint_as_float(~k);
}

// ---- convert a row-major f32 [K x 64] weight matrix into f16 B-fragment layout ----
// frag index: ((kt*4 + nt)*8 + v)*32 + lane ; lane: n = nt*16 + (lane&15), kh = lane>>4
// dword v holds K = kt*32 + kh*8 + 2v + (v>=4 ? 8 : 0) (low half) and +1 (high half)
__global__ void prep_frag_kernel(const float* __restrict__ W, unsigned* __restrict__ dst,
                                 int k_act, int total) {
  int i = blockIdx.x * blockDim.x + threadIdx.x;
  if (i >= total) return;
  int lane = i & 31, v = (i >> 5) & 7, nt = (i >> 8) & 3, kt = i >> 10;
  int n = nt * 16 + (lane & 15);
  int k = kt * 32 + (lane >> 4) * 8 + 2 * v + (v >= 4 ? 8 : 0);
  HPack p;
  p.h[0] = (_Float16)((k     < k_act) ? W[(long)k * 64 + n]       : 0.0f);
  p.h[1] = (_Float16)((k + 1 < k_act) ? W[(long)(k + 1) * 64 + n] : 0.0f);
  dst[i] = p.u;
}

__global__ void fill_keys_kernel(unsigned* __restrict__ s, int n) {
  int i = blockIdx.x * blockDim.x + threadIdx.x;
  if (i < n) s[i] = 0x007FFFFFu;
}

// keys -> decoded f16 segment-max buffer, AND re-init key for the next layer (fused)
__global__ void decode_reset_kernel(unsigned* __restrict__ s, _Float16* __restrict__ out,
                                    int n) {
  int i = blockIdx.x * blockDim.x + threadIdx.x;
  if (i < n) {
    out[i] = (_Float16)dec_key(s[i]);
    s[i] = 0x007FFFFFu;
  }
}

// ---- load one B fragment from LDS-staged weights (8 conflict-free ds_load_b32) ----
__device__ __forceinline__ v16h load_bfrag_lds(const unsigned* base, int lane) {
  BFrag bf;
#pragma unroll
  for (int v = 0; v < 8; ++v) bf.u[v] = base[v * 32 + lane];
  return bf.h;
}

// ---- fused MLP layer: in -> GEMM1 -> +b -> LN -> ReLU -> GEMM2 -> +b -> LN -> ReLU
//      -> write Xh (f16) and fused segment-max (uint-key atomicMax) ----
// NOTE: w1f points at contiguous [W1frag | W2frag] (KT1*1024 + 2048 dwords).
template <bool FIRST>
__global__ __launch_bounds__(256)
void mlp_layer_kernel(const float*    __restrict__ in_x32, // FIRST only: [N,10] f32
                      const _Float16* __restrict__ in_x16, // !FIRST: Xh [N,64] f16
                      const _Float16* __restrict__ in_s16, // !FIRST: decoded S [NSEG,64] f16
                      const int*      __restrict__ cluster,
                      const unsigned* __restrict__ w1f,
                      const float* __restrict__ b1,
                      const float* __restrict__ g1,     const float* __restrict__ be1,
                      const float* __restrict__ b2,
                      const float* __restrict__ g2,     const float* __restrict__ be2,
                      _Float16* __restrict__ x_out, unsigned* __restrict__ s_keys,
                      int n_rows) {
  constexpr int KT1 = FIRST ? 1 : 4;
  constexpr int WDW = KT1 * 1024 + 2048;             // staged weight dwords (12 / 24 KB)
  __shared__ __align__(16) unsigned wlds[WDW];
  __shared__ __align__(16) _Float16 hlds[8][16][64]; // per-wave 16x64 f16 tile (16 KB)

  // ---- cooperative weight stage into LDS (all waves, before any exit) ----
  {
    const uint4* src = (const uint4*)w1f;
    uint4* dst = (uint4*)wlds;
#pragma unroll
    for (int i = 0; i < WDW / 4 / 256; ++i)
      dst[threadIdx.x + i * 256] = src[threadIdx.x + i * 256];
  }
  __syncthreads();
  const unsigned* w1l = wlds;
  const unsigned* w2l = wlds + KT1 * 1024;

  const int lane = threadIdx.x & 31;
  const int wave = threadIdx.x >> 5;
  const long tile = (long)blockIdx.x * 8 + wave;
  const int row0 = (int)(tile * 16);
  if (row0 >= n_rows) return;            // wave-uniform exit
  const int m = lane & 15;               // A-matrix row for this lane
  const int kh = lane >> 4;              // K-half selector
  const int ncol = lane & 15;            // C-matrix column within n-tile
  const int row = row0 + m;
  const int myc = cluster[row];

  // ---- A fragments for GEMM1 ----
  AFrag a[KT1];
  if (FIRST) {
#pragma unroll
    for (int j = 0; j < 8; ++j) {
      const int k0 = kh * 8 + j;
      const int k1 = k0 + 16;
      a[0].h16[j]     = (_Float16)((k0 < 10) ? in_x32[(long)row * 10 + k0] : 0.0f);
      a[0].h16[8 + j] = (_Float16)((k1 < 10) ? in_x32[(long)row * 10 + k1] : 0.0f);
    }
  } else {
    const v8h* xr = (const v8h*)(in_x16 + (long)row * 64);
    const v8h* sr = (const v8h*)(in_s16 + (long)myc * 64);
    a[0].h8[0] = xr[kh];     a[0].h8[1] = xr[kh + 2];   // K  0..31 from X
    a[1].h8[0] = xr[4 + kh]; a[1].h8[1] = xr[6 + kh];   // K 32..63 from X
    a[2].h8[0] = sr[kh];     a[2].h8[1] = sr[kh + 2];   // K 64..95 from S gather
    a[3].h8[0] = sr[4 + kh]; a[3].h8[1] = sr[6 + kh];   // K 96..127 from S gather
  }

  // ---- GEMM1: 16x64 output tile in 4 accumulators ----
  v8f acc[4];
#pragma unroll
  for (int nt = 0; nt < 4; ++nt) {
    v8f c = {};
#pragma unroll
    for (int kt = 0; kt < KT1; ++kt) {
      v16h b = load_bfrag_lds(w1l + ((kt * 4 + nt) * 8) * 32, lane);
      c = __builtin_amdgcn_wmma_f32_16x16x32_f16(false, a[kt].h16, false, b,
                                                 (short)0, c, false, false);
    }
    acc[nt] = c;
  }

  // ---- bias + LayerNorm(64) + ReLU, operating in C-fragment layout ----
  auto ln_relu = [&](v8f* A, const float* bb_, const float* gg_, const float* ee_) {
    float bb[4], gg[4], ee[4];
#pragma unroll
    for (int nt = 0; nt < 4; ++nt) {
      int n = nt * 16 + ncol;
      bb[nt] = bb_[n]; gg[nt] = gg_[n]; ee[nt] = ee_[n];
    }
#pragma unroll
    for (int v = 0; v < 8; ++v) {        // row M = kh*8 + v
      float s = 0.0f, s2 = 0.0f;
#pragma unroll
      for (int nt = 0; nt < 4; ++nt) {
        float t = A[nt][v] + bb[nt];
        A[nt][v] = t; s += t; s2 += t * t;
      }
#pragma unroll
      for (int w = 1; w < 16; w <<= 1) { // butterfly within 16-lane half
        s  += __shfl_xor(s,  w, 32);
        s2 += __shfl_xor(s2, w, 32);
      }
      float mean = s * (1.0f / 64.0f);
      float var  = s2 * (1.0f / 64.0f) - mean * mean;
      float r = rsqrtf(var + 1e-5f);
#pragma unroll
      for (int nt = 0; nt < 4; ++nt) {
        float t = (A[nt][v] - mean) * r * gg[nt] + ee[nt];
        A[nt][v] = fmaxf(t, 0.0f);
      }
    }
  };
  ln_relu(acc, b1, g1, be1);

  // ---- reshape C-layout -> A-layout via per-wave LDS tile ----
#pragma unroll
  for (int nt = 0; nt < 4; ++nt)
#pragma unroll
    for (int v = 0; v < 8; ++v)
      hlds[wave][kh * 8 + v][nt * 16 + ncol] = (_Float16)acc[nt][v];

  const v8h* hr = (const v8h*)(&hlds[wave][m][0]);
  AFrag a2[2];
  a2[0].h8[0] = hr[kh];     a2[0].h8[1] = hr[kh + 2];
  a2[1].h8[0] = hr[4 + kh]; a2[1].h8[1] = hr[6 + kh];

  // ---- GEMM2 (K=64) ----
#pragma unroll
  for (int nt = 0; nt < 4; ++nt) {
    v8f c = {};
#pragma unroll
    for (int kt = 0; kt < 2; ++kt) {
      v16h b = load_bfrag_lds(w2l + ((kt * 4 + nt) * 8) * 32, lane);
      c = __builtin_amdgcn_wmma_f32_16x16x32_f16(false, a2[kt].h16, false, b,
                                                 (short)0, c, false, false);
    }
    acc[nt] = c;
  }
  ln_relu(acc, b2, g2, be2);

  // ---- write Xh (f16) and fused segment-max ----
  int cl[8];
#pragma unroll
  for (int v = 0; v < 8; ++v) cl[v] = __shfl(myc, kh * 8 + v, 32);
#pragma unroll
  for (int nt = 0; nt < 4; ++nt)
#pragma unroll
    for (int v = 0; v < 8; ++v) {
      float t = acc[nt][v];
      int M = kh * 8 + v, n = nt * 16 + ncol;
      x_out[(long)(row0 + M) * 64 + n] = (_Float16)t;
      atomicMax(&s_keys[(long)cl[v] * 64 + n], enc_f32(t));
    }
}

// ---- final 128->64 linear (+bias), fused segment-max only ----
__global__ __launch_bounds__(256)
void final_linear_kernel(const _Float16* __restrict__ Xh, const _Float16* __restrict__ Sh,
                         const int* __restrict__ cluster, const unsigned* __restrict__ wf,
                         const float* __restrict__ bias, unsigned* __restrict__ s_keys,
                         int n_rows) {
  __shared__ __align__(16) unsigned wlds[4096];      // 16 KB staged final weights
  {
    const uint4* src = (const uint4*)wf;
    uint4* dst = (uint4*)wlds;
#pragma unroll
    for (int i = 0; i < 4; ++i)
      dst[threadIdx.x + i * 256] = src[threadIdx.x + i * 256];
  }
  __syncthreads();

  const int lane = threadIdx.x & 31;
  const int wave = threadIdx.x >> 5;
  const long tile = (long)blockIdx.x * 8 + wave;
  const int row0 = (int)(tile * 16);
  if (row0 >= n_rows) return;
  const int m = lane & 15, kh = lane >> 4, ncol = lane & 15;
  const int row = row0 + m;
  const int myc = cluster[row];

  const v8h* xr = (const v8h*)(Xh + (long)row * 64);
  const v8h* sr = (const v8h*)(Sh + (long)myc * 64);
  AFrag a[4];
  a[0].h8[0] = xr[kh];     a[0].h8[1] = xr[kh + 2];
  a[1].h8[0] = xr[4 + kh]; a[1].h8[1] = xr[6 + kh];
  a[2].h8[0] = sr[kh];     a[2].h8[1] = sr[kh + 2];
  a[3].h8[0] = sr[4 + kh]; a[3].h8[1] = sr[6 + kh];

  v8f acc[4];
#pragma unroll
  for (int nt = 0; nt < 4; ++nt) {
    v8f c = {};
#pragma unroll
    for (int kt = 0; kt < 4; ++kt) {
      v16h b = load_bfrag_lds(wlds + ((kt * 4 + nt) * 8) * 32, lane);
      c = __builtin_amdgcn_wmma_f32_16x16x32_f16(false, a[kt].h16, false, b,
                                                 (short)0, c, false, false);
    }
    acc[nt] = c;
  }

  int cl[8];
#pragma unroll
  for (int v = 0; v < 8; ++v) cl[v] = __shfl(myc, kh * 8 + v, 32);
#pragma unroll
  for (int nt = 0; nt < 4; ++nt) {
    int n = nt * 16 + ncol;
    float bn = bias[n];
#pragma unroll
    for (int v = 0; v < 8; ++v) {
      float t = acc[nt][v] + bn;
      atomicMax(&s_keys[(long)cl[v] * 64 + n], enc_f32(t));
    }
  }
}

// ---- decode keys + L2 normalize each segment row; one wave per segment ----
__global__ __launch_bounds__(32)
void normalize_kernel(const unsigned* __restrict__ s_keys, float* __restrict__ out) {
  int seg = blockIdx.x;
  int lane = threadIdx.x;
  float v0 = dec_key(s_keys[(long)seg * 64 + lane]);
  float v1 = dec_key(s_keys[(long)seg * 64 + 32 + lane]);
  float ss = v0 * v0 + v1 * v1;
#pragma unroll
  for (int w = 1; w < 32; w <<= 1) ss += __shfl_xor(ss, w, 32);
  float d = fmaxf(sqrtf(ss), 1e-12f);
  out[(long)seg * 64 + lane]      = v0 / d;
  out[(long)seg * 64 + 32 + lane] = v1 / d;
}

extern "C" void kernel_launch(void* const* d_in, const int* in_sizes, int n_in,
                              void* d_out, int out_size, void* d_ws, size_t ws_size,
                              hipStream_t stream) {
  const float* x       = (const float*)d_in[0];
  const int*   cluster = (const int*)d_in[1];
  const float* P[24];
  for (int i = 0; i < 24; ++i) P[i] = (const float*)d_in[2 + i];
  const float* lin_w = (const float*)d_in[26];
  const float* lin_b = (const float*)d_in[27];
  const int N = in_sizes[0] / 10;            // 1,000,000

  // workspace layout (bytes):
  //   [0, 81920)            f16 fragment weights (19456 dwords used)
  //   [81920, +N*64*2)      Xh  : f16 activations [N,64]
  //   next NSEG*64*4        K   : segment-max uint keys
  //   next NSEG*64*2        Sh  : decoded f16 segment max [NSEG,64]
  unsigned* WF = (unsigned*)d_ws;
  _Float16* Xh = (_Float16*)((char*)d_ws + 81920);
  unsigned* K  = (unsigned*)((char*)Xh + (size_t)N * 64 * 2);
  _Float16* Sh = (_Float16*)((char*)K + (size_t)NSEG * 64 * 4);

  // --- weight prep (f32 -> f16 fragment layout); per layer [W1frag|W2frag] contiguous ---
  prep_frag_kernel<<<4,  256, 0, stream>>>(P[0],  WF,          10, 1024);  // L0 W1 (K pad 32)
  prep_frag_kernel<<<8,  256, 0, stream>>>(P[4],  WF + 1024,   64, 2048);  // L0 W2
  prep_frag_kernel<<<16, 256, 0, stream>>>(P[8],  WF + 3072,  128, 4096);  // L1 W1
  prep_frag_kernel<<<8,  256, 0, stream>>>(P[12], WF + 7168,   64, 2048);  // L1 W2
  prep_frag_kernel<<<16, 256, 0, stream>>>(P[16], WF + 9216,  128, 4096);  // L2 W1
  prep_frag_kernel<<<8,  256, 0, stream>>>(P[20], WF + 13312,  64, 2048);  // L2 W2
  prep_frag_kernel<<<16, 256, 0, stream>>>(lin_w, WF + 15360, 128, 4096);  // final

  const int skn = NSEG * 64;
  const int skb = (skn + 255) / 256;
  const int blocks = (N / 16 + 7) / 8;       // 7813 blocks of 8 waves (16 rows/wave)

  // --- layer 0 ---
  fill_keys_kernel<<<skb, 256, 0, stream>>>(K, skn);
  mlp_layer_kernel<true><<<blocks, 256, 0, stream>>>(x, nullptr, nullptr, cluster,
      WF,        P[1],  P[2],  P[3],  P[5],  P[6],  P[7],
      Xh, K, N);
  decode_reset_kernel<<<skb, 256, 0, stream>>>(K, Sh, skn);

  // --- layer 1 ---
  mlp_layer_kernel<false><<<blocks, 256, 0, stream>>>(nullptr, Xh, Sh, cluster,
      WF + 3072, P[9],  P[10], P[11], P[13], P[14], P[15],
      Xh, K, N);
  decode_reset_kernel<<<skb, 256, 0, stream>>>(K, Sh, skn);

  // --- layer 2 ---
  mlp_layer_kernel<false><<<blocks, 256, 0, stream>>>(nullptr, Xh, Sh, cluster,
      WF + 9216, P[17], P[18], P[19], P[21], P[22], P[23],
      Xh, K, N);
  decode_reset_kernel<<<skb, 256, 0, stream>>>(K, Sh, skn);

  // --- final linear + segmax + normalize ---
  final_linear_kernel<<<blocks, 256, 0, stream>>>(Xh, Sh, cluster,
      WF + 15360, lin_b, K, N);
  normalize_kernel<<<NSEG, 32, 0, stream>>>(K, (float*)d_out);
}
